// ViT_48790828483105
// MI455X (gfx1250) — matrix operand order repaired and checked by reading, not compiled, not run on hardware
//
// ViT forward for MI455X (gfx1250, wave32, WMMA bf16).
//
// Key optimization: the reference's attention uses softmax only through its
// row-sum (==1), so the attention tensor collapses to a per-head constant
// c_aw[h] = LN_g(colsum(re_w))*ren_w+ren_b, and
//   y[b,h,i,d] = c_aw[h] * sum_j rope_inv(v)[b,h,j,d]   (constant over i).
// Q/K projections, q/k layernorm, QK^T and softmax are dead code. The
// remaining hot path is GEMMs (V-proj, fc1, fc2) run as bf16 WMMA with f32
// accumulation: 128x128x32 block tiles, 8 waves x (2x4) v_wmma_f32_16x16x32_bf16,
// double-buffered LDS so global fetch of tile k+1 overlaps WMMA on tile k.
//
// Workspace requirement: ~126 MB (6 * T*E + T*4E floats + small tails).

#include <hip/hip_runtime.h>
#include <hip/hip_bf16.h>
#include <math.h>

#define B_   16
#define N_   256
#define T_   (B_*N_)     // 4096 tokens
#define E_   768
#define H_   12
#define D_   64
#define FE_  3072        // 4*E mlp hidden
#define EPS_ 1e-5f

typedef __bf16 bf16_t;
typedef __attribute__((ext_vector_type(16))) __bf16 v16bf;
typedef __attribute__((ext_vector_type(8)))  __bf16 v8bf;
typedef __attribute__((ext_vector_type(4)))  __bf16 v4bf;
typedef __attribute__((ext_vector_type(8)))  float  v8f;

union Frag16 { v16bf v; v8bf h[2]; };

__device__ __forceinline__ float gelu_exact(float x) {
    return 0.5f * x * (1.f + erff(x * 0.70710678118654752440f));
}

// ---------------------------------------------------------------------------
// bf16 WMMA GEMM:  C[M x N] = act( A[M x K] @ W[N x K]^T + bias[N] )
// M % 128 == 0, N % 128 == 0, K % 32 == 0. 256 threads (8 wave32 waves).
// Each wave computes a 32x64 sub-tile as 2x4 WMMA 16x16 accumulators.
// Double-buffered LDS: global loads of the next K tile are issued before the
// WMMAs of the current tile; fp32->bf16 conversion + ds_store happen after.
// ---------------------------------------------------------------------------
__global__ __launch_bounds__(256)
void gemm_bf16_wmma(const float* __restrict__ A, const float* __restrict__ W,
                    const float* __restrict__ bias, float* __restrict__ C,
                    int M, int N, int K, int act)
{
    constexpr int LDSW = 40;                 // bf16 elems per LDS row (32 + 8 pad -> 80B stride)
    __shared__ __align__(16) bf16_t As[2][128 * LDSW];
    __shared__ __align__(16) bf16_t Bs[2][128 * LDSW];

    const int tid  = threadIdx.x;
    const int lane = tid & 31;
    const int wave = tid >> 5;
    const int wm   = wave & 3;               // 4 wave rows  -> 32 M-rows each
    const int wn   = wave >> 2;              // 2 wave cols  -> 64 N-cols each
    const int bm   = blockIdx.y * 128;
    const int bn   = blockIdx.x * 128;

    const int quad  = tid & 7;               // which float4 within a 32-float row
    const int rbase = tid >> 3;              // 0..31
    const int r16   = lane & 15;
    const int hl    = lane >> 4;             // half-wave select

    // per-thread global row pointers (4 rows each, 32 rows apart), hoisted
    const float* pA[4]; const float* pB[4];
    #pragma unroll
    for (int q = 0; q < 4; ++q) {
        pA[q] = A + (size_t)(bm + rbase + 32 * q) * K + quad * 4;
        pB[q] = W + (size_t)(bn + rbase + 32 * q) * K + quad * 4;
    }

    v8f acc[2][4];
    #pragma unroll
    for (int mi = 0; mi < 2; ++mi)
        #pragma unroll
        for (int ni = 0; ni < 4; ++ni)
            #pragma unroll
            for (int r = 0; r < 8; ++r) acc[mi][ni][r] = 0.f;

    // ---- prologue: fetch + stage tile 0 ----
    float4 ga[4], gb[4];
    #pragma unroll
    for (int q = 0; q < 4; ++q) { ga[q] = *(const float4*)pA[q]; gb[q] = *(const float4*)pB[q]; }
    #pragma unroll
    for (int q = 0; q < 4; ++q) {
        const int r = rbase + 32 * q;
        v4bf ca; ca[0]=(bf16_t)ga[q].x; ca[1]=(bf16_t)ga[q].y; ca[2]=(bf16_t)ga[q].z; ca[3]=(bf16_t)ga[q].w;
        *(v4bf*)&As[0][r * LDSW + quad * 4] = ca;
        v4bf cb; cb[0]=(bf16_t)gb[q].x; cb[1]=(bf16_t)gb[q].y; cb[2]=(bf16_t)gb[q].z; cb[3]=(bf16_t)gb[q].w;
        *(v4bf*)&Bs[0][r * LDSW + quad * 4] = cb;
    }

    const int nk = K >> 5;
    for (int kt = 0; kt < nk; ++kt) {
        __syncthreads();                     // staged data visible; prev reads of nxt buf done
        const int cur = kt & 1, nxt = cur ^ 1;
        const bool more = (kt + 1) < nk;

        // ---- issue global loads for tile kt+1 (latency hidden behind WMMAs) ----
        if (more) {
            const int ko = (kt + 1) << 5;
            #pragma unroll
            for (int q = 0; q < 4; ++q) { ga[q] = *(const float4*)(pA[q] + ko); gb[q] = *(const float4*)(pB[q] + ko); }
        }

        // ---- fragments per the CDNA5 16-bit A/B VGPR layouts + 8 WMMAs ----
        Frag16 fa[2], fb[4];
        #pragma unroll
        for (int mi = 0; mi < 2; ++mi) {
            const int row = wm * 32 + mi * 16 + r16;
            // A 16x32: half-wave 0 holds K 0-7 & 16-23; half-wave 1 holds K 8-15 & 24-31
            fa[mi].h[0] = *(const v8bf*)&As[cur][row * LDSW + 8 * hl];
            fa[mi].h[1] = *(const v8bf*)&As[cur][row * LDSW + 16 + 8 * hl];
        }
        #pragma unroll
        for (int ni = 0; ni < 4; ++ni) {
            const int col = wn * 64 + ni * 16 + r16;
            // B 32x16: half-wave 0 holds K 0-7 & 16-23; half-wave 1 holds K 8-15 & 24-31
            fb[ni].h[0] = *(const v8bf*)&Bs[cur][col * LDSW + 8 * hl];
            fb[ni].h[1] = *(const v8bf*)&Bs[cur][col * LDSW + 16 + 8 * hl];
        }
        #pragma unroll
        for (int mi = 0; mi < 2; ++mi)
            #pragma unroll
            for (int ni = 0; ni < 4; ++ni)
                acc[mi][ni] = __builtin_amdgcn_wmma_f32_16x16x32_bf16(
                    false, fa[mi].v, false, fb[ni].v, (short)0, acc[mi][ni], false, false);

        // ---- convert + stage tile kt+1 into the other buffer ----
        if (more) {
            #pragma unroll
            for (int q = 0; q < 4; ++q) {
                const int r = rbase + 32 * q;
                v4bf ca; ca[0]=(bf16_t)ga[q].x; ca[1]=(bf16_t)ga[q].y; ca[2]=(bf16_t)ga[q].z; ca[3]=(bf16_t)ga[q].w;
                *(v4bf*)&As[nxt][r * LDSW + quad * 4] = ca;
                v4bf cb; cb[0]=(bf16_t)gb[q].x; cb[1]=(bf16_t)gb[q].y; cb[2]=(bf16_t)gb[q].z; cb[3]=(bf16_t)gb[q].w;
                *(v4bf*)&Bs[nxt][r * LDSW + quad * 4] = cb;
            }
        }
    }

    // ---- epilogue: bias + optional GELU ----
    #pragma unroll
    for (int mi = 0; mi < 2; ++mi) {
        #pragma unroll
        for (int ni = 0; ni < 4; ++ni) {
            const int col = bn + wn * 64 + ni * 16 + r16;
            const float bv = bias ? bias[col] : 0.f;
            #pragma unroll
            for (int r = 0; r < 8; ++r) {
                const int row = bm + wm * 32 + mi * 16 + r + hl * 8;
                float v = acc[mi][ni][r] + bv;
                if (act == 1) v = gelu_exact(v);
                C[(size_t)row * N + col] = v;
            }
        }
    }
}

// ---------------------------------------------------------------------------
// Stem convolutions (2% of FLOPs -> plain fp32 VALU kernels)
// ---------------------------------------------------------------------------
__global__ void stem_conv1(const float* __restrict__ x, const float* __restrict__ w,
                           float* __restrict__ out)
{
    // x (16,3,256,256), w (192,3,4,4) stride 4 -> out (16,192,64,64)
    const int idx = blockIdx.x * 256 + threadIdx.x;
    const int px = idx & 63, py = (idx >> 6) & 63;
    const int co = (idx >> 12) % 192, b = idx / (4096 * 192);
    const float* xin = x + (size_t)b * 3 * 256 * 256;
    const float* wp  = w + (size_t)co * 48;
    float acc = 0.f;
    for (int ci = 0; ci < 3; ++ci)
        #pragma unroll
        for (int ky = 0; ky < 4; ++ky)
            #pragma unroll
            for (int kx = 0; kx < 4; ++kx)
                acc += xin[((size_t)ci * 256 + (py * 4 + ky)) * 256 + px * 4 + kx]
                     * wp[(ci * 4 + ky) * 4 + kx];
    out[idx] = acc;
}

__global__ void stem_conv2x2(const float* __restrict__ x, const float* __restrict__ w,
                             float* __restrict__ out, int Cin, int Cout,
                             int Ho, int Wo, int tokens)
{
    const int idx = blockIdx.x * 256 + threadIdx.x;
    const int px = idx % Wo, py = (idx / Wo) % Ho;
    const int co = (idx / (Wo * Ho)) % Cout, b = idx / (Wo * Ho * Cout);
    const int Hi = Ho * 2, Wi = Wo * 2;
    const float* xin = x + (size_t)b * Cin * Hi * Wi;
    const float* wp  = w + (size_t)co * Cin * 4;
    float acc = 0.f;
    for (int ci = 0; ci < Cin; ++ci) {
        const float* xp = xin + ((size_t)ci * Hi + py * 2) * Wi + px * 2;
        const float* wq = wp + ci * 4;
        acc += xp[0] * wq[0] + xp[1] * wq[1] + xp[Wi] * wq[2] + xp[Wi + 1] * wq[3];
    }
    if (tokens)  // conv3: write directly in (b, n, e) token layout
        out[((size_t)b * N_ + py * Wo + px) * E_ + co] = acc;
    else
        out[idx] = acc;
}

__global__ void instnorm_hw(float* __restrict__ d, const float* __restrict__ g,
                            const float* __restrict__ be, int C, int HW, int apply_gelu)
{
    const int bc = blockIdx.x, c = bc % C;
    float* p = d + (size_t)bc * HW;
    __shared__ float s1[256], s2[256];
    float a = 0.f, q = 0.f;
    for (int i = threadIdx.x; i < HW; i += 256) { float v = p[i]; a += v; q += v * v; }
    s1[threadIdx.x] = a; s2[threadIdx.x] = q; __syncthreads();
    for (int st = 128; st > 0; st >>= 1) {
        if (threadIdx.x < st) { s1[threadIdx.x] += s1[threadIdx.x + st]; s2[threadIdx.x] += s2[threadIdx.x + st]; }
        __syncthreads();
    }
    const float m = s1[0] / HW, var = s2[0] / HW - m * m;
    const float inv = rsqrtf(var + EPS_), gg = g[c], bb = be[c];
    for (int i = threadIdx.x; i < HW; i += 256) {
        float v = (p[i] - m) * inv * gg + bb;
        if (apply_gelu) v = gelu_exact(v);
        p[i] = v;
    }
}

__global__ void instnorm_tok(float* __restrict__ d, const float* __restrict__ g,
                             const float* __restrict__ be)
{
    // per (b, c) over N_=256 tokens, data in (b*N + n)*E + c layout
    const int b = blockIdx.x / E_, c = blockIdx.x % E_;
    float* p = d + (size_t)b * N_ * E_ + c;
    __shared__ float s1[256], s2[256];
    const float v = p[(size_t)threadIdx.x * E_];
    s1[threadIdx.x] = v; s2[threadIdx.x] = v * v; __syncthreads();
    for (int st = 128; st > 0; st >>= 1) {
        if (threadIdx.x < st) { s1[threadIdx.x] += s1[threadIdx.x + st]; s2[threadIdx.x] += s2[threadIdx.x + st]; }
        __syncthreads();
    }
    const float m = s1[0] * (1.f / N_), var = s2[0] * (1.f / N_) - m * m;
    p[(size_t)threadIdx.x * E_] = (v - m) * rsqrtf(var + EPS_) * g[c] + be[c];
}

// ---------------------------------------------------------------------------
// Row layernorm over E_=768, 256 threads, 3 elements / thread
// ---------------------------------------------------------------------------
__global__ void layernorm_row(const float* __restrict__ x, const float* __restrict__ w,
                              const float* __restrict__ b, float* __restrict__ out)
{
    __shared__ float s1[256], s2[256];
    const int row = blockIdx.x, t = threadIdx.x;
    const float* p = x + (size_t)row * E_;
    const float v0 = p[t], v1 = p[t + 256], v2 = p[t + 512];
    s1[t] = v0 + v1 + v2; s2[t] = v0 * v0 + v1 * v1 + v2 * v2;
    __syncthreads();
    for (int st = 128; st > 0; st >>= 1) {
        if (t < st) { s1[t] += s1[t + st]; s2[t] += s2[t + st]; }
        __syncthreads();
    }
    const float m = s1[0] * (1.f / E_), var = s2[0] * (1.f / E_) - m * m;
    const float inv = rsqrtf(var + EPS_);
    float* o = out + (size_t)row * E_;
    o[t]       = (v0 - m) * inv * w[t]       + b[t];
    o[t + 256] = (v1 - m) * inv * w[t + 256] + b[t + 256];
    o[t + 512] = (v2 - m) * inv * w[t + 512] + b[t + 512];
}

// c_aw[g] = LN_g( sum_h re_w[h,g] ) * ren_w[g] + ren_b[g]
__global__ void compute_aw(const float* __restrict__ re_w, const float* __restrict__ rw,
                           const float* __restrict__ rb, float* __restrict__ caw)
{
    __shared__ float cs[H_]; __shared__ float mv[2];
    const int t = threadIdx.x;
    if (t < H_) { float s = 0.f; for (int h = 0; h < H_; ++h) s += re_w[h * H_ + t]; cs[t] = s; }
    __syncthreads();
    if (t == 0) {
        float m = 0.f; for (int g = 0; g < H_; ++g) m += cs[g];
        m *= (1.f / H_);
        float v = 0.f; for (int g = 0; g < H_; ++g) { float d = cs[g] - m; v += d * d; }
        v *= (1.f / H_);
        mv[0] = m; mv[1] = rsqrtf(v + EPS_);
    }
    __syncthreads();
    if (t < H_) caw[t] = (cs[t] - mv[0]) * mv[1] * rw[t] + rb[t];
}

// y_const[b, h*64+d] = c_aw[h] * sum_j rope_inv(V)[b,h,j,d]
__global__ void rope_reduce(const float* __restrict__ V, const float* __restrict__ caw,
                            float* __restrict__ y)
{
    const int h = blockIdx.x % H_, b = blockIdx.x / H_;
    const int d = threadIdx.x;                       // 64 threads
    const float* vp = V + (size_t)b * N_ * E_ + h * D_;
    float acc = 0.f;
    if (d < 32) {                                    // rotary half (rot_dim = 32)
        const int i = d >> 1, odd = d & 1;
        const float invf = powf(10000.f, -(float)(2 * i) / 32.f);
        const float bs   = ((float)(2 * i) + 0.4f * 32.f) / (1.4f * 32.f);
        const float lbs  = logf(bs);
        for (int j = 0; j < N_; ++j) {
            float sph, cph; sincosf((float)j * invf, &sph, &cph);
            const float sv = expf(-lbs * ((float)(j - N_ / 2) * (1.f / 512.f))); // 1/scale
            const float e0 = vp[(size_t)j * E_ + 2 * i];
            const float e1 = vp[(size_t)j * E_ + 2 * i + 1];
            acc += sv * (odd ? (e1 * cph + e0 * sph) : (e0 * cph - e1 * sph));
        }
    } else {                                         // pass-through half
        for (int j = 0; j < N_; ++j) acc += vp[(size_t)j * E_ + d];
    }
    y[(size_t)b * E_ + h * D_ + d] = caw[h] * acc;
}

// out[b,e] = proj_b[e] + sum_k y_const[b,k] * proj_w[e,k]   (16 x 768 matvec batch)
__global__ void proj_small(const float* __restrict__ y, const float* __restrict__ pw,
                           const float* __restrict__ pb, float* __restrict__ out)
{
    const int b = blockIdx.x, e = blockIdx.y * 64 + threadIdx.x;
    const float* yv = y + (size_t)b * E_;
    const float* wr = pw + (size_t)e * E_;
    float acc = pb[e];
    for (int k = 0; k < E_; ++k) acc += yv[k] * wr[k];
    out[(size_t)b * E_ + e] = acc;
}

// x[t,e] += ls1[e] * (A0[b,e] + A1[b,e])   (attention output is constant over tokens)
__global__ void add_attn(float* __restrict__ x, const float* __restrict__ ls,
                         const float* __restrict__ a0, const float* __restrict__ a1)
{
    const int idx = blockIdx.x * 256 + threadIdx.x;
    const int e = idx % E_, b = idx / (N_ * E_);
    x[idx] += ls[e] * (a0[(size_t)b * E_ + e] + a1[(size_t)b * E_ + e]);
}

__global__ void add_mlp(float* __restrict__ x, const float* __restrict__ ls,
                        const float* __restrict__ m0, const float* __restrict__ m1)
{
    const int idx = blockIdx.x * 256 + threadIdx.x;
    const int e = idx % E_;
    x[idx] += ls[e] * (m0[idx] + m1[idx]);
}

// out = 2*x + dwconv3x3(x) + peg_b   (h = h + peg(h) and peg returns conv+b+h)
__global__ void peg_kernel(const float* __restrict__ x, const float* __restrict__ pw,
                           const float* __restrict__ pb, float* __restrict__ out)
{
    const int idx = blockIdx.x * 256 + threadIdx.x;
    const int e = idx % E_, t = idx / E_, b = t / N_, n = t % N_;
    const int py = n >> 4, px = n & 15;
    const float* xb = x + (size_t)b * N_ * E_ + e;
    float acc = 0.f;
    #pragma unroll
    for (int dy = -1; dy <= 1; ++dy) {
        const int yy = py + dy; if (yy < 0 || yy > 15) continue;
        #pragma unroll
        for (int dx = -1; dx <= 1; ++dx) {
            const int xx = px + dx; if (xx < 0 || xx > 15) continue;
            acc += pw[e * 9 + (dy + 1) * 3 + (dx + 1)] * xb[(size_t)(yy * 16 + xx) * E_];
        }
    }
    out[idx] = 2.f * x[idx] + acc + pb[e];
}

// ---------------------------------------------------------------------------
extern "C" void kernel_launch(void* const* d_in, const int* in_sizes, int n_in,
                              void* d_out, int out_size, void* d_ws, size_t ws_size,
                              hipStream_t stream)
{
    (void)in_sizes; (void)n_in; (void)out_size; (void)ws_size;
    auto F = [&](int i) { return (const float*)d_in[i]; };

    // ---- workspace layout (floats) ----
    const size_t SZ_TE = (size_t)T_ * E_;    // 3,145,728
    const size_t SZ_TF = (size_t)T_ * FE_;   // 12,582,912
    float* XA = (float*)d_ws;
    float* XB = XA + SZ_TE;                  // PEG ping-pong
    float* HB = XB + SZ_TE;                  // ln1/ln2 output (also stem mid buffer)
    float* H2 = HB + SZ_TE;                  // per-mlp inner LN output
    float* VB = H2 + SZ_TE;                  // V projection / mlp0 output
    float* M1 = VB + SZ_TE;                  // mlp1 output
    float* G  = M1 + SZ_TE;                  // fc1 output T x 3072 (also stem conv1 out)
    float* YC = G + SZ_TF;                   // y_const B x E
    float* A0 = YC + (size_t)B_ * E_;
    float* A1 = A0 + (size_t)B_ * E_;
    float* CAW = A1 + (size_t)B_ * E_;       // 2 x 16

    // ---- stem: conv(4x4 s4) -> IN+gelu -> conv(2x2 s2) -> IN+gelu -> conv -> IN ----
    stem_conv1<<<(B_ * 192 * 64 * 64) / 256, 256, 0, stream>>>(F(0), F(1), G);
    instnorm_hw<<<B_ * 192, 256, 0, stream>>>(G, F(2), F(3), 192, 64 * 64, 1);
    stem_conv2x2<<<(B_ * 192 * 32 * 32) / 256, 256, 0, stream>>>(G, F(4), HB, 192, 192, 32, 32, 0);
    instnorm_hw<<<B_ * 192, 256, 0, stream>>>(HB, F(5), F(6), 192, 32 * 32, 1);
    stem_conv2x2<<<(B_ * 768 * 16 * 16) / 256, 256, 0, stream>>>(HB, F(7), XA, 192, 768, 16, 16, 1);
    instnorm_tok<<<B_ * E_, 256, 0, stream>>>(XA, F(8), F(9));

    float* x = XA; float* xalt = XB;
    for (int blk = 0; blk < 6; ++blk) {
        const int bb = 12 + 40 * blk;        // flat-input base of this block's params
        // ---- attention (collapsed) ----
        layernorm_row<<<T_, 256, 0, stream>>>(x, F(bb + 0), F(bb + 1), HB);
        for (int p = 0; p < 2; ++p) {
            const int ab = bb + 6 + 11 * p;  // qkv_w,qkv_b,proj_w,proj_b,qn_w,qn_b,kn_w,kn_b,re_w,ren_w,ren_b
            compute_aw<<<1, 32, 0, stream>>>(F(ab + 8), F(ab + 9), F(ab + 10), CAW + 16 * p);
            const float* Wv = F(ab + 0) + (size_t)2 * E_ * E_;   // V rows of qkv_w
            const float* bv = F(ab + 1) + 2 * E_;
            dim3 gv(E_ / 128, T_ / 128);
            gemm_bf16_wmma<<<gv, 256, 0, stream>>>(HB, Wv, bv, VB, T_, E_, E_, 0);
            rope_reduce<<<B_ * H_, 64, 0, stream>>>(VB, CAW + 16 * p, YC);
            dim3 gp(B_, E_ / 64);
            proj_small<<<gp, 64, 0, stream>>>(YC, F(ab + 2), F(ab + 3), p == 0 ? A0 : A1);
        }
        add_attn<<<(T_ * E_) / 256, 256, 0, stream>>>(x, F(bb + 4), A0, A1);

        if (blk == 0) {                      // PEG after block 0
            peg_kernel<<<(T_ * E_) / 256, 256, 0, stream>>>(x, F(10), F(11), xalt);
            float* t = x; x = xalt; xalt = t;
        }

        // ---- MLPs ----
        layernorm_row<<<T_, 256, 0, stream>>>(x, F(bb + 2), F(bb + 3), HB);
        for (int p = 0; p < 2; ++p) {
            const int mb = bb + 28 + 6 * p;  // ln_w,ln_b,fc1_w,fc1_b,fc2_w,fc2_b
            layernorm_row<<<T_, 256, 0, stream>>>(HB, F(mb + 0), F(mb + 1), H2);
            dim3 g1(FE_ / 128, T_ / 128);
            gemm_bf16_wmma<<<g1, 256, 0, stream>>>(H2, F(mb + 2), F(mb + 3), G, T_, FE_, E_, 1);
            dim3 g2(E_ / 128, T_ / 128);
            gemm_bf16_wmma<<<g2, 256, 0, stream>>>(G, F(mb + 4), F(mb + 5),
                                                   p == 0 ? VB : M1, T_, E_, FE_, 0);
        }
        add_mlp<<<(T_ * E_) / 256, 256, 0, stream>>>(x, F(bb + 5), VB, M1);
    }

    // ---- final layernorm -> d_out ----
    layernorm_row<<<T_, 256, 0, stream>>>(x, F(252), F(253), (float*)d_out);
}
